// Speller_69724499083547
// MI455X (gfx1250) — matrix operand order repaired and checked by reading, not compiled, not run on hardware
//
#include <hip/hip_runtime.h>
#include <hip/hip_bf16.h>

#define B   64
#define TE  1000
#define TD  300
#define Dm  512
#define V   31

typedef __attribute__((ext_vector_type(16))) _Float16 v16h;
typedef __attribute__((ext_vector_type(8)))  _Float16 v8h;
typedef __attribute__((ext_vector_type(8)))  float    v8f;

// ---------------- WMMA fragment loaders (gfx1250 wave32 layouts) ----------------
// A 16x32 f16: lane L: row = m0 + (L&15); halves 0..7 = K kb..kb+7, halves 8..15 = K kb+16..kb+23, kb = 8*(L>>4)
__device__ __forceinline__ v16h load_a_h(const _Float16* __restrict__ A, int ld, int m0, int k0, int lane) {
    int r  = m0 + (lane & 15);
    int kb = k0 + ((lane >> 4) << 3);
    const _Float16* p = A + (size_t)r * ld + kb;
    v8h lo = *(const v8h*)p;
    v8h hi = *(const v8h*)(p + 16);
    v16h a;
#pragma unroll
    for (int i = 0; i < 8; ++i) { a[i] = lo[i]; a[i + 8] = hi[i]; }
    return a;
}

// Same A fragment but source is f32 row-major (convert on load)
__device__ __forceinline__ v16h load_a_f(const float* __restrict__ A, int ld, int m0, int k0, int lane) {
    int r  = m0 + (lane & 15);
    int kb = k0 + ((lane >> 4) << 3);
    const float* p = A + (size_t)r * ld + kb;
    v16h a;
#pragma unroll
    for (int i = 0; i < 8; ++i) { a[i] = (_Float16)p[i]; a[i + 8] = (_Float16)p[i + 16]; }
    return a;
}

// B 32x16 f16 with B[k][n] = W[n0+n][k0+k] (W row-major, contraction along W rows):
// lane L: col = n0 + (L&15); halves 0..7 = K kb..kb+7, halves 8..15 = K kb+16..kb+23, kb = 8*(L>>4)
__device__ __forceinline__ v16h load_b_h(const _Float16* __restrict__ W, int ld, int n0, int k0, int lane) {
    int n  = n0 + (lane & 15);
    int kb = k0 + ((lane >> 4) << 3);
    const _Float16* p = W + (size_t)n * ld + kb;
    v8h lo = *(const v8h*)p;
    v8h hi = *(const v8h*)(p + 16);
    v16h b;
#pragma unroll
    for (int i = 0; i < 8; ++i) { b[i] = lo[i]; b[i + 8] = hi[i]; }
    return b;
}

// ---------------- precompute kernels ----------------
__global__ void init_k(float* c0, float* c1, _Float16* h0H, _Float16* h1H, _Float16* ctxH, int* prev) {
    int i = blockIdx.x * blockDim.x + threadIdx.x;
    if (i < B * Dm) {
        c0[i] = 0.f; c1[i] = 0.f;
        h0H[i] = (_Float16)0.f; h1H[i] = (_Float16)0.f; ctxH[i] = (_Float16)0.f;
    }
    if (i < B) prev[i] = 0; // SOS
}

__global__ void cvt_k(const float* __restrict__ s, _Float16* __restrict__ d, int n) {
    int i = blockIdx.x * blockDim.x + threadIdx.x;
    if (i < n) d[i] = (_Float16)s[i];
}

// d[c*R + r] = s[r*C + c]   (transpose + convert)
__global__ void cvtT_k(const float* __restrict__ s, _Float16* __restrict__ d, int R, int C) {
    int i = blockIdx.x * blockDim.x + threadIdx.x;
    if (i < R * C) {
        int r = i / C, c = i % C;
        d[(size_t)c * R + r] = (_Float16)s[i];
    }
}

__global__ void emb_pad_k(const float* __restrict__ emb, _Float16* __restrict__ embH) {
    int i = blockIdx.x * blockDim.x + threadIdx.x; // 32*512
    int r = i >> 9, dd = i & 511;
    embH[i] = (r < V) ? (_Float16)emb[r * Dm + dd] : (_Float16)0.f;
}

// keys/vals: Out[r][n] = sum_k Enc[r][k] * WT[n][k]   (Enc f32 64000x512, WT f16 512x512)
__global__ void enc_gemm(const float* __restrict__ Enc, const _Float16* __restrict__ WT,
                         _Float16* __restrict__ Out) {
    int wave = blockIdx.x * (blockDim.x >> 5) + (threadIdx.x >> 5);
    int lane = threadIdx.x & 31;
    int m0 = (wave >> 5) << 4;   // 4000 m-tiles
    int n0 = (wave & 31) << 4;   // 32 n-tiles
    v8f acc = {};
    for (int k0 = 0; k0 < Dm; k0 += 32) {
        v16h a = load_a_f(Enc, Dm, m0, k0, lane);
        v16h b = load_b_h(WT, Dm, n0, k0, lane);
        acc = __builtin_amdgcn_wmma_f32_16x16x32_f16(false, a, false, b, (short)0, acc, false, false);
    }
    int n = n0 + (lane & 15);
    int rb = m0 + ((lane >> 4) << 3);
#pragma unroll
    for (int v = 0; v < 8; ++v) Out[(size_t)(rb + v) * Dm + n] = (_Float16)acc[v];
}

// ---------------- per-step kernels ----------------
__global__ void embed_k(const int* __restrict__ y, const int* __restrict__ prev,
                        const float* __restrict__ emb, const _Float16* __restrict__ ctxH,
                        _Float16* __restrict__ xcatH, int t) {
    int i = blockIdx.x * blockDim.x + threadIdx.x; // 64*512
    int b = i >> 9, d = i & 511;
    int sym = (t > 0 && t < TD - 1) ? y[b * TD + t] : prev[b];
    xcatH[b * 2 * Dm + d]       = (_Float16)emb[sym * Dm + d];
    xcatH[b * 2 * Dm + Dm + d]  = ctxH[b * Dm + d];
}

// G[64][2048] = X(64xK1) @ Wih^T + H(64xK2) @ Whh^T + bih + bhh
__global__ void gates_gemm(const _Float16* __restrict__ X, int ldx, int K1,
                           const _Float16* __restrict__ Wih,
                           const _Float16* __restrict__ H, int K2,
                           const _Float16* __restrict__ Whh,
                           const float* __restrict__ bih, const float* __restrict__ bhh,
                           float* __restrict__ G) {
    const int N = 4 * Dm;
    int wave = blockIdx.x * (blockDim.x >> 5) + (threadIdx.x >> 5);
    int lane = threadIdx.x & 31;
    int nt = N >> 4;                      // 128
    int m0 = (wave / nt) << 4;
    int n0 = (wave % nt) << 4;
    v8f acc = {};
    for (int k0 = 0; k0 < K1; k0 += 32) {
        v16h a = load_a_h(X, ldx, m0, k0, lane);
        v16h b = load_b_h(Wih, K1, n0, k0, lane);
        acc = __builtin_amdgcn_wmma_f32_16x16x32_f16(false, a, false, b, (short)0, acc, false, false);
    }
    for (int k0 = 0; k0 < K2; k0 += 32) {
        v16h a = load_a_h(H, K2, m0, k0, lane);
        v16h b = load_b_h(Whh, K2, n0, k0, lane);
        acc = __builtin_amdgcn_wmma_f32_16x16x32_f16(false, a, false, b, (short)0, acc, false, false);
    }
    int n = n0 + (lane & 15);
    float bias = bih[n] + bhh[n];
    int rb = m0 + ((lane >> 4) << 3);
#pragma unroll
    for (int v = 0; v < 8; ++v) G[(size_t)(rb + v) * N + n] = acc[v] + bias;
}

__global__ void cell_k(const float* __restrict__ G, float* __restrict__ c, _Float16* __restrict__ hH) {
    int idx = blockIdx.x * blockDim.x + threadIdx.x; // 64*512
    int b = idx >> 9, j = idx & 511;
    const float* g = G + (size_t)b * 4 * Dm;
    float gi = g[j], gf = g[Dm + j], gg = g[2 * Dm + j], go = g[3 * Dm + j];
    float si = 1.f / (1.f + __expf(-gi));
    float sf = 1.f / (1.f + __expf(-gf));
    float so = 1.f / (1.f + __expf(-go));
    float c2 = sf * c[idx] + si * tanhf(gg);
    c[idx] = c2;
    hH[idx] = (_Float16)(so * tanhf(c2));
}

// O(64xN f16) = relu?( X(64xK) @ W^T + bias )
__global__ void gemm_xwT_h(const _Float16* __restrict__ X, int ldx,
                           const _Float16* __restrict__ W, int K,
                           const float* __restrict__ bias, int do_relu,
                           _Float16* __restrict__ O, int N) {
    int wave = blockIdx.x * (blockDim.x >> 5) + (threadIdx.x >> 5);
    int lane = threadIdx.x & 31;
    int nt = N >> 4;
    int m0 = (wave / nt) << 4;
    int n0 = (wave % nt) << 4;
    v8f acc = {};
    for (int k0 = 0; k0 < K; k0 += 32) {
        v16h a = load_a_h(X, ldx, m0, k0, lane);
        v16h b = load_b_h(W, K, n0, k0, lane);
        acc = __builtin_amdgcn_wmma_f32_16x16x32_f16(false, a, false, b, (short)0, acc, false, false);
    }
    int n = n0 + (lane & 15);
    float bb = bias ? bias[n] : 0.f;
    int rb = m0 + ((lane >> 4) << 3);
#pragma unroll
    for (int v = 0; v < 8; ++v) {
        float x = acc[v] + bb;
        if (do_relu) x = fmaxf(x, 0.f);
        O[(size_t)(rb + v) * N + n] = (_Float16)x;
    }
}

__device__ __forceinline__ float wave_max(float v) {
#pragma unroll
    for (int o = 16; o; o >>= 1) v = fmaxf(v, __shfl_xor(v, o, 32));
    return v;
}
__device__ __forceinline__ float wave_sum(float v) {
#pragma unroll
    for (int o = 16; o; o >>= 1) v += __shfl_xor(v, o, 32);
    return v;
}

// e = (qH . keysH)/sqrt(D), mask, softmax over TE; writes attention_plot row
__global__ void attn_k(const _Float16* __restrict__ qH, const _Float16* __restrict__ keysH,
                       const int* __restrict__ enc_len, float* __restrict__ att, int t) {
    int b = blockIdx.x;
    int tid = threadIdx.x;            // 1024
    int lane = tid & 31, wid = tid >> 5;
    __shared__ float qs[Dm];
    __shared__ float redm[32];
    __shared__ float reds[32];
    if (tid < Dm) qs[tid] = (float)qH[b * Dm + tid];
    __syncthreads();

    float myE = -1e9f;
    if (tid < TE) {
        const _Float16* kp = keysH + ((size_t)b * TE + tid) * Dm;
        float s = 0.f;
        for (int d0 = 0; d0 < Dm; d0 += 8) {
            v8h kv = *(const v8h*)(kp + d0);
#pragma unroll
            for (int i = 0; i < 8; ++i) s += qs[d0 + i] * (float)kv[i];
        }
        myE = (tid < enc_len[b]) ? s * 0.04419417382415922f : -1e9f;
    }
    float m = wave_max(myE);
    if (lane == 0) redm[wid] = m;
    __syncthreads();
    if (tid < 32) { float x = redm[tid]; x = wave_max(x); if (tid == 0) redm[0] = x; }
    __syncthreads();
    float bmax = redm[0];

    float w = (tid < TE) ? __expf(myE - bmax) : 0.f;
    float sm = wave_sum(w);
    if (lane == 0) reds[wid] = sm;
    __syncthreads();
    if (tid < 32) { float x = reds[tid]; x = wave_sum(x); if (tid == 0) reds[0] = x; }
    __syncthreads();
    float inv = 1.f / reds[0];

    if (tid < TE) att[((size_t)b * TD + t) * TE + tid] = w * inv;
}

// ctx = w @ vals ; build cdnH=[h1|ctx] f16 and ctxH
__global__ void ctx_k(const float* __restrict__ att, const _Float16* __restrict__ valsH,
                      const _Float16* __restrict__ h1H, _Float16* __restrict__ cdnH,
                      _Float16* __restrict__ ctxH, int t) {
    int b = blockIdx.x;
    int d = threadIdx.x;              // 512
    __shared__ float ws_[TE];
    for (int i = d; i < TE; i += Dm) ws_[i] = att[((size_t)b * TD + t) * TE + i];
    __syncthreads();
    float acc = 0.f;
    const _Float16* vp = valsH + (size_t)b * TE * Dm + d;
    for (int tt = 0; tt < TE; ++tt) acc += ws_[tt] * (float)vp[(size_t)tt * Dm];
    _Float16 hc = h1H[b * Dm + d];
    cdnH[b * 2 * Dm + d]      = hc;
    cdnH[b * 2 * Dm + Dm + d] = (_Float16)acc;
    ctxH[b * Dm + d]          = (_Float16)acc;
}

// logits = hidH @ embH^T + b_char  -> d_out[b][t][0..30]
__global__ void logits_k(const _Float16* __restrict__ hidH, const _Float16* __restrict__ embH,
                         const float* __restrict__ bchar, float* __restrict__ outL, int t) {
    int wave = threadIdx.x >> 5;     // 8 waves: 4 m-tiles x 2 n-tiles
    int lane = threadIdx.x & 31;
    int m0 = (wave >> 1) << 4;
    int n0 = (wave & 1) << 4;
    v8f acc = {};
    for (int k0 = 0; k0 < Dm; k0 += 32) {
        v16h a = load_a_h(hidH, Dm, m0, k0, lane);
        v16h b = load_b_h(embH, Dm, n0, k0, lane);
        acc = __builtin_amdgcn_wmma_f32_16x16x32_f16(false, a, false, b, (short)0, acc, false, false);
    }
    int n = n0 + (lane & 15);
    if (n < V) {
        float bb = bchar[n];
        int rb = m0 + ((lane >> 4) << 3);
#pragma unroll
        for (int v = 0; v < 8; ++v)
            outL[((size_t)(rb + v) * TD + t) * V + n] = acc[v] + bb;
    }
}

__global__ void argmax_k(const float* __restrict__ outL, int* __restrict__ prev, int t) {
    int b = threadIdx.x; // 64
    const float* p = outL + ((size_t)b * TD + t) * V;
    float best = p[0]; int bi = 0;
    for (int v = 1; v < V; ++v) { float x = p[v]; if (x > best) { best = x; bi = v; } }
    prev[b] = bi;
}

// ---------------- workspace layout (bytes) ----------------
#define OFF_KEYS  ((size_t)0)
#define OFF_VALS  (OFF_KEYS + (size_t)B*TE*Dm*2)
#define OFF_WIH0  (OFF_VALS + (size_t)B*TE*Dm*2)
#define OFF_WHH0  (OFF_WIH0 + (size_t)4*Dm*2*Dm*2)
#define OFF_WIH1  (OFF_WHH0 + (size_t)4*Dm*Dm*2)
#define OFF_WHH1  (OFF_WIH1 + (size_t)4*Dm*Dm*2)
#define OFF_WQT   (OFF_WHH1 + (size_t)4*Dm*Dm*2)
#define OFF_WKT   (OFF_WQT  + (size_t)Dm*Dm*2)
#define OFF_WVT   (OFF_WKT  + (size_t)Dm*Dm*2)
#define OFF_WOUT  (OFF_WVT  + (size_t)Dm*Dm*2)
#define OFF_EMBH  (OFF_WOUT + (size_t)Dm*2*Dm*2)
#define OFF_XCAT  (OFF_EMBH + (size_t)32*Dm*2)
#define OFF_H0H   (OFF_XCAT + (size_t)B*2*Dm*2)
#define OFF_H1H   (OFF_H0H  + (size_t)B*Dm*2)
#define OFF_QH    (OFF_H1H  + (size_t)B*Dm*2)
#define OFF_HIDH  (OFF_QH   + (size_t)B*Dm*2)
#define OFF_CDNH  (OFF_HIDH + (size_t)B*Dm*2)
#define OFF_CTXH  (OFF_CDNH + (size_t)B*2*Dm*2)
#define OFF_C0    (OFF_CTXH + (size_t)B*Dm*2)
#define OFF_C1    (OFF_C0   + (size_t)B*Dm*4)
#define OFF_GATE  (OFF_C1   + (size_t)B*Dm*4)
#define OFF_PREV  (OFF_GATE + (size_t)B*4*Dm*4)

extern "C" void kernel_launch(void* const* d_in, const int* in_sizes, int n_in,
                              void* d_out, int out_size, void* d_ws, size_t ws_size,
                              hipStream_t stream) {
    const float* enc    = (const float*)d_in[0];
    const int*   elen   = (const int*)d_in[1];
    const int*   y      = (const int*)d_in[2];
    const float* emb    = (const float*)d_in[3];
    const float* Wih0   = (const float*)d_in[4];
    const float* Whh0   = (const float*)d_in[5];
    const float* bih0   = (const float*)d_in[6];
    const float* bhh0   = (const float*)d_in[7];
    const float* Wih1   = (const float*)d_in[8];
    const float* Whh1   = (const float*)d_in[9];
    const float* bih1   = (const float*)d_in[10];
    const float* bhh1   = (const float*)d_in[11];
    const float* Wq     = (const float*)d_in[12];
    const float* Wk     = (const float*)d_in[13];
    const float* Wv     = (const float*)d_in[14];
    const float* Wout   = (const float*)d_in[15];
    const float* b_out  = (const float*)d_in[16];
    const float* b_char = (const float*)d_in[17];

    char* ws = (char*)d_ws;
    _Float16* keysH = (_Float16*)(ws + OFF_KEYS);
    _Float16* valsH = (_Float16*)(ws + OFF_VALS);
    _Float16* Wih0H = (_Float16*)(ws + OFF_WIH0);
    _Float16* Whh0H = (_Float16*)(ws + OFF_WHH0);
    _Float16* Wih1H = (_Float16*)(ws + OFF_WIH1);
    _Float16* Whh1H = (_Float16*)(ws + OFF_WHH1);
    _Float16* WqTH  = (_Float16*)(ws + OFF_WQT);
    _Float16* WkTH  = (_Float16*)(ws + OFF_WKT);
    _Float16* WvTH  = (_Float16*)(ws + OFF_WVT);
    _Float16* WoutH = (_Float16*)(ws + OFF_WOUT);
    _Float16* embH  = (_Float16*)(ws + OFF_EMBH);
    _Float16* xcatH = (_Float16*)(ws + OFF_XCAT);
    _Float16* h0H   = (_Float16*)(ws + OFF_H0H);
    _Float16* h1H   = (_Float16*)(ws + OFF_H1H);
    _Float16* qH    = (_Float16*)(ws + OFF_QH);
    _Float16* hidH  = (_Float16*)(ws + OFF_HIDH);
    _Float16* cdnH  = (_Float16*)(ws + OFF_CDNH);
    _Float16* ctxH  = (_Float16*)(ws + OFF_CTXH);
    float*    c0    = (float*)(ws + OFF_C0);
    float*    c1    = (float*)(ws + OFF_C1);
    float*    gates = (float*)(ws + OFF_GATE);
    int*      prev  = (int*)(ws + OFF_PREV);

    float* outL = (float*)d_out;                       // [B][TD][31]
    float* outA = outL + (size_t)B * TD * V;           // [B][TD][TE]

    // ---- precompute: state init, f16 weights (Wq/Wk/Wv transposed), keys/vals GEMMs ----
    init_k<<<(B * Dm + 255) / 256, 256, 0, stream>>>(c0, c1, h0H, h1H, ctxH, prev);
    cvt_k<<<(4 * Dm * 2 * Dm + 255) / 256, 256, 0, stream>>>(Wih0, Wih0H, 4 * Dm * 2 * Dm);
    cvt_k<<<(4 * Dm * Dm + 255) / 256, 256, 0, stream>>>(Whh0, Whh0H, 4 * Dm * Dm);
    cvt_k<<<(4 * Dm * Dm + 255) / 256, 256, 0, stream>>>(Wih1, Wih1H, 4 * Dm * Dm);
    cvt_k<<<(4 * Dm * Dm + 255) / 256, 256, 0, stream>>>(Whh1, Whh1H, 4 * Dm * Dm);
    cvt_k<<<(Dm * 2 * Dm + 255) / 256, 256, 0, stream>>>(Wout, WoutH, Dm * 2 * Dm);
    cvtT_k<<<(Dm * Dm + 255) / 256, 256, 0, stream>>>(Wq, WqTH, Dm, Dm);
    cvtT_k<<<(Dm * Dm + 255) / 256, 256, 0, stream>>>(Wk, WkTH, Dm, Dm);
    cvtT_k<<<(Dm * Dm + 255) / 256, 256, 0, stream>>>(Wv, WvTH, Dm, Dm);
    emb_pad_k<<<(32 * Dm + 255) / 256, 256, 0, stream>>>(emb, embH);
    // 64000x512 rows, 32 n-tiles -> 128000 waves, 8 waves/block
    enc_gemm<<<16000, 256, 0, stream>>>(enc, WkTH, keysH);
    enc_gemm<<<16000, 256, 0, stream>>>(enc, WvTH, valsH);

    // ---- 300 sequential decode steps ----
    for (int t = 0; t < TD; ++t) {
        embed_k<<<(B * Dm + 255) / 256, 256, 0, stream>>>(y, prev, emb, ctxH, xcatH, t);
        gates_gemm<<<64, 256, 0, stream>>>(xcatH, 2 * Dm, 2 * Dm, Wih0H, h0H, Dm, Whh0H,
                                           bih0, bhh0, gates);
        cell_k<<<(B * Dm + 255) / 256, 256, 0, stream>>>(gates, c0, h0H);
        gates_gemm<<<64, 256, 0, stream>>>(h0H, Dm, Dm, Wih1H, h1H, Dm, Whh1H,
                                           bih1, bhh1, gates);
        cell_k<<<(B * Dm + 255) / 256, 256, 0, stream>>>(gates, c1, h1H);
        gemm_xwT_h<<<16, 256, 0, stream>>>(h1H, Dm, WqTH, Dm, nullptr, 0, qH, Dm);
        attn_k<<<B, 1024, 0, stream>>>(qH, keysH, elen, outA, t);
        ctx_k<<<B, Dm, 0, stream>>>(outA, valsH, h1H, cdnH, ctxH, t);
        gemm_xwT_h<<<16, 256, 0, stream>>>(cdnH, 2 * Dm, WoutH, 2 * Dm, b_out, 1, hidH, Dm);
        logits_k<<<1, 256, 0, stream>>>(hidH, embH, b_char, outL, t);
        argmax_k<<<1, 64, 0, stream>>>(outL, prev, t);
    }
}
